// GemmaAttention_9363028706284
// MI455X (gfx1250) — matrix operand order repaired and checked by reading, not compiled, not run on hardware
//
#include <hip/hip_runtime.h>
#include <hip/hip_bf16.h>
#include <math.h>

// ---------------------------------------------------------------------------
// GemmaAttention on MI455X (gfx1250).
//   B=2 S=2048 HID=2048 H=8 KV=1 D=256  (fp32 in / fp32 out)
// Compute-bound (~146 GFLOP vs ~100 MB): everything GEMM-shaped goes through
// v_wmma_f32_16x16x32_bf16; weights/activations are pre-converted to bf16 so
// WMMA operands stream as 16B vector loads; GEMM weight tiles are staged into
// LDS with global_load_async_to_lds_b128 (ASYNCcnt) and double-buffered.
// ---------------------------------------------------------------------------

#define B_   2
#define S_   2048
#define HID_ 2048
#define H_   8
#define KV_  1
#define D_   256
#define T_   (B_ * S_)          // 4096 tokens

typedef __attribute__((ext_vector_type(16))) __bf16 v16bf;
typedef __attribute__((ext_vector_type(8)))  float  v8f;
typedef __attribute__((ext_vector_type(8)))  __bf16 bf16x8;
typedef __attribute__((ext_vector_type(2)))  __bf16 bf16x2;

// Build a 16-element bf16 A/B fragment from two contiguous 8-element segments.
__device__ __forceinline__ v16bf frag_cat(bf16x8 lo, bf16x8 hi) {
  v16bf f;
#pragma unroll
  for (int e = 0; e < 8; ++e) { f[e] = lo[e]; f[8 + e] = hi[e]; }
  return f;
}

// ---------------------------------------------------------------------------
// fp32 -> bf16 pre-conversion (packed, emits v_cvt_pk_bf16_f32)
// ---------------------------------------------------------------------------
__global__ __launch_bounds__(256) void cvt_f32_bf16(
    const float* __restrict__ in, __bf16* __restrict__ out, int n) {
  const int i = (blockIdx.x * blockDim.x + threadIdx.x) * 2;
  if (i >= n) return;
  float2 f = *(const float2*)(in + i);
  bf16x2 o;
  o.x = (__bf16)f.x; o.y = (__bf16)f.y;
  *(bf16x2*)(out + i) = o;
}

// ---------------------------------------------------------------------------
// GEMM: C[T x N] = A[T x K] * W[N x K]^T  (bf16 in, bf16 or fp32 out)
// 256-thread block = 8 waves stacked in M (block tile 128 x 16).
// The shared 16 x 32 weight tile is staged into LDS with
// global_load_async_to_lds_b128, double-buffered, synced via s_wait_asynccnt.
// ---------------------------------------------------------------------------
__global__ __launch_bounds__(256) void gemm_bf16_wmma(
    const __bf16* __restrict__ A, const __bf16* __restrict__ W,
    void* __restrict__ C, int T, int N, int K, int storeBf) {
  // 16 rows x 32 bf16, padded to 40 (80B row stride: 16B-aligned segments,
  // conflict-free across 16 lanes on 64 banks), double buffered.
  __shared__ __attribute__((aligned(16))) __bf16 wt[2][16][40];

  const int lane = threadIdx.x & 31;
  const int wave = threadIdx.x >> 5;
  const int half = lane >> 4;
  const int ln   = lane & 15;

  const int mblocks = T >> 7;                 // 128 rows per block
  const int bm = blockIdx.x % mblocks;
  const int bn = blockIdx.x / mblocks;
  const int tm = bm * 128 + wave * 16;
  const int tn = bn * 16;

  const __bf16* ap = A + (size_t)(tm + ln) * K;   // this lane's A row

  // Issue one 16x32 bf16 weight tile (1KB) as 64 async b128 transfers,
  // spread over all 8 waves (8 per wave -> ASYNCcnt <= 16 when 2 in flight).
  auto issue_tile = [&](int t, int buf) {
    if ((threadIdx.x & 3) == 0) {
      const int c   = threadIdx.x >> 2;       // 0..63
      const int row = c >> 2;                 // 0..15
      const int seg = c & 3;                  // 16B segment within the row
      unsigned ldsOff = (unsigned)(uintptr_t)&wt[buf][row][seg * 8];
      unsigned gOff   = (unsigned)((((size_t)(tn + row) * K) + (size_t)t * 32) * 2
                                   + seg * 16);
      asm volatile("global_load_async_to_lds_b128 %0, %1, %2"
                   :: "v"(ldsOff), "v"(gOff), "s"(W)
                   : "memory");
    }
  };

  const int nT = K >> 5;                      // K-tiles of 32
  issue_tile(0, 0);

  v8f acc = {};
  for (int it = 0; it < nT; ++it) {
    const int cur = it & 1;
    const bool hasNext = (it + 1) < nT;
    if (hasNext) issue_tile(it + 1, cur ^ 1);

    // Async loads complete in order: <=8 outstanding means current tile done.
    if (hasNext) asm volatile("s_wait_asynccnt 0x8" ::: "memory");
    else         asm volatile("s_wait_asynccnt 0x0" ::: "memory");
    __syncthreads();                          // all waves' portions visible

    const int kb = it * 32;
    __builtin_prefetch(ap + kb + 256, 0, 1);

    // A fragment (16x32, interleaved-half layout = two contiguous 16B runs)
    bf16x8 a0 = *(const bf16x8*)(ap + kb + 8 * half);
    bf16x8 a1 = *(const bf16x8*)(ap + kb + 16 + 8 * half);
    // B fragment from LDS (same interleaved-half layout as A)
    bf16x8 b0 = *(const bf16x8*)&wt[cur][ln][8 * half];
    bf16x8 b1 = *(const bf16x8*)&wt[cur][ln][16 + 8 * half];

    acc = __builtin_amdgcn_wmma_f32_16x16x32_bf16(
        false, frag_cat(a0, a1), false, frag_cat(b0, b1), (short)0, acc,
        false, false);

    __syncthreads();                          // safe to overwrite buffer cur
  }

  // C/D layout: VGPR r = row r (lanes 0-15) / r+8 (lanes 16-31); col = ln.
  const int crow = tm + 8 * half;
  if (storeBf) {
    __bf16* Cb = (__bf16*)C;
#pragma unroll
    for (int r = 0; r < 8; ++r)
      Cb[(size_t)(crow + r) * N + tn + ln] = (__bf16)acc[r];
  } else {
    float* Cf = (float*)C;
#pragma unroll
    for (int r = 0; r < 8; ++r)
      Cf[(size_t)(crow + r) * N + tn + ln] = acc[r];
  }
}

// ---------------------------------------------------------------------------
// RoPE applied in place to bf16 q [T, H*D] and k [T, KV*D].
// ---------------------------------------------------------------------------
__global__ __launch_bounds__(256) void rope_kernel(
    __bf16* __restrict__ q, __bf16* __restrict__ kbuf,
    const int* __restrict__ pos) {
  const int idx   = blockIdx.x * blockDim.x + threadIdx.x;
  const int total = T_ * (H_ + KV_) * (D_ / 2);
  if (idx >= total) return;
  const int i    = idx % (D_ / 2);
  const int rem  = idx / (D_ / 2);
  const int head = rem % (H_ + KV_);
  const int t    = rem / (H_ + KV_);

  // inv_freq[i] = 10000^(-2i/256) = exp(-i * ln(10000)/128)
  const float inv   = __expf(-(float)i * 0.07195578f);
  const float angle = (float)pos[t] * inv;
  float s, c;
  __sincosf(angle, &s, &c);

  __bf16* p = (head < H_) ? (q + (size_t)t * (H_ * D_) + head * D_)
                          : (kbuf + (size_t)t * D_);
  const float x1 = (float)p[i];
  const float x2 = (float)p[i + D_ / 2];
  p[i]          = (__bf16)(x1 * c - x2 * s);
  p[i + D_ / 2] = (__bf16)(x2 * c + x1 * s);
}

// ---------------------------------------------------------------------------
// Flash attention: grid = B*H*(S/16), 128 threads (4 waves) per block.
// Waves share the 16x32 WMMA score tile + online softmax; each wave owns a
// 64-wide slice of D for P*V. P transposes C-layout -> A-layout through LDS.
// ---------------------------------------------------------------------------
__global__ __launch_bounds__(128) void attn_kernel(
    const __bf16* __restrict__ qmat,   // [T, H*D]
    const __bf16* __restrict__ kmat,   // [T, D]
    const __bf16* __restrict__ vmat,   // [T, D]
    const float*  __restrict__ mask,   // [B, 1, S, S]
    __bf16* __restrict__ omat) {       // [T, H*D] (bf16 for final WMMA GEMM)
  __shared__ __attribute__((aligned(16))) __bf16 pls[4][16][40];

  const int lane = threadIdx.x & 31;
  const int wv   = threadIdx.x >> 5;   // 0..3 -> 64-wide slice of D
  const int half = lane >> 4;
  const int ln   = lane & 15;

  const int qt = blockIdx.x % (S_ / 16);
  const int bh = blockIdx.x / (S_ / 16);
  const int h  = bh % H_;
  const int b  = bh / H_;
  const int q0 = qt * 16;
  const int dbase = wv * 64;

  // Preload Q tile (16 x 256) as 8 bf16 A-fragments (two 16B loads each).
  const __bf16* qbase =
      qmat + (size_t)(b * S_ + q0 + ln) * (H_ * D_) + h * D_;
  v16bf qf[8];
#pragma unroll
  for (int c = 0; c < 8; ++c) {
    bf16x8 lo = *(const bf16x8*)(qbase + 32 * c + 8 * half);
    bf16x8 hi = *(const bf16x8*)(qbase + 32 * c + 16 + 8 * half);
    qf[c] = frag_cat(lo, hi);
  }

  v8f accd[4] = {{}, {}, {}, {}};
  float mrun[8], lrun[8];
#pragma unroll
  for (int r = 0; r < 8; ++r) { mrun[r] = -3.0e38f; lrun[r] = 0.0f; }

  for (int kt = 0; kt < S_; kt += 32) {
    // ---- scores S[16x32] = Q . K^T, accumulated over 32-d chunks
    v8f s0 = {}, s1 = {};
    const __bf16* kb0 = kmat + (size_t)(b * S_ + kt + ln) * D_;
    const __bf16* kb1 = kb0 + (size_t)16 * D_;
#pragma unroll
    for (int c = 0; c < 8; ++c) {
      v16bf kf0 = frag_cat(*(const bf16x8*)(kb0 + 32 * c + 8 * half),
                           *(const bf16x8*)(kb0 + 32 * c + 16 + 8 * half));
      v16bf kf1 = frag_cat(*(const bf16x8*)(kb1 + 32 * c + 8 * half),
                           *(const bf16x8*)(kb1 + 32 * c + 16 + 8 * half));
      s0 = __builtin_amdgcn_wmma_f32_16x16x32_bf16(false, qf[c], false, kf0,
                                                   (short)0, s0, false, false);
      s1 = __builtin_amdgcn_wmma_f32_16x16x32_bf16(false, qf[c], false, kf1,
                                                   (short)0, s1, false, false);
    }

    // ---- online softmax across this 32-key tile
    float scale[8];
#pragma unroll
    for (int r = 0; r < 8; ++r) {
      const size_t mrow = ((size_t)b * S_ + q0 + r + 8 * half) * S_ + kt;
      const float sv0 = s0[r] * 0.0625f + mask[mrow + ln];       // 1/sqrt(256)
      const float sv1 = s1[r] * 0.0625f + mask[mrow + 16 + ln];
      float mx = fmaxf(sv0, sv1);
      mx = fmaxf(mx, __shfl_xor(mx, 1, 32));
      mx = fmaxf(mx, __shfl_xor(mx, 2, 32));
      mx = fmaxf(mx, __shfl_xor(mx, 4, 32));
      mx = fmaxf(mx, __shfl_xor(mx, 8, 32));
      const float nm  = fmaxf(mrun[r], mx);
      const float sc  = __expf(mrun[r] - nm);
      const float p0v = __expf(sv0 - nm);
      const float p1v = __expf(sv1 - nm);
      float rs = p0v + p1v;
      rs += __shfl_xor(rs, 1, 32);
      rs += __shfl_xor(rs, 2, 32);
      rs += __shfl_xor(rs, 4, 32);
      rs += __shfl_xor(rs, 8, 32);
      lrun[r]  = lrun[r] * sc + rs;
      mrun[r]  = nm;
      scale[r] = sc;
      pls[wv][r + 8 * half][ln]      = (__bf16)p0v;   // C-layout -> LDS
      pls[wv][r + 8 * half][16 + ln] = (__bf16)p1v;
    }
#pragma unroll
    for (int c2 = 0; c2 < 4; ++c2)
#pragma unroll
      for (int r = 0; r < 8; ++r) accd[c2][r] *= scale[r];

    asm volatile("s_wait_dscnt 0x0" ::: "memory");    // wave-local transpose

    // ---- P back as 16x32 A-fragment (two 16B LDS reads)
    v16bf pf = frag_cat(*(const bf16x8*)&pls[wv][ln][16 * half],
                        *(const bf16x8*)&pls[wv][ln][16 * half + 8]);

    // ---- out += P (16x32) . V (32 x 64-slice)
#pragma unroll
    for (int c2 = 0; c2 < 4; ++c2) {
      v16bf vf;
      const __bf16* vp =
          vmat + (size_t)(b * S_ + kt + 16 * half) * D_ + dbase + 16 * c2 + ln;
#pragma unroll
      for (int e = 0; e < 16; ++e) vf[e] = vp[(size_t)e * D_];
      accd[c2] = __builtin_amdgcn_wmma_f32_16x16x32_bf16(
          false, pf, false, vf, (short)0, accd[c2], false, false);
    }
  }

  // ---- epilogue: normalize, store bf16 [T, H*D]
#pragma unroll
  for (int c2 = 0; c2 < 4; ++c2) {
#pragma unroll
    for (int r = 0; r < 8; ++r) {
      const int row = q0 + r + 8 * half;
      const int col = h * D_ + dbase + 16 * c2 + ln;
      omat[(size_t)(b * S_ + row) * (H_ * D_) + col] =
          (__bf16)(accd[c2][r] / lrun[r]);
    }
  }
}

// ---------------------------------------------------------------------------
extern "C" void kernel_launch(void* const* d_in, const int* in_sizes, int n_in,
                              void* d_out, int out_size, void* d_ws,
                              size_t ws_size, hipStream_t stream) {
  const float* hidden = (const float*)d_in[0];
  const float* mask   = (const float*)d_in[1];
  const int*   pos    = (const int*)d_in[2];
  const float* wq     = (const float*)d_in[3];
  const float* wk     = (const float*)d_in[4];
  const float* wv     = (const float*)d_in[5];
  const float* wo     = (const float*)d_in[6];
  float*       out    = (float*)d_out;

  // bf16 workspace layout
  __bf16* hb  = (__bf16*)d_ws;                        // T*HID
  __bf16* wqb = hb  + (size_t)T_ * HID_;              // (H*D)*HID
  __bf16* wkb = wqb + (size_t)H_ * D_ * HID_;         // (KV*D)*HID
  __bf16* wvb = wkb + (size_t)KV_ * D_ * HID_;
  __bf16* wob = wvb + (size_t)KV_ * D_ * HID_;        // HID*(H*D)
  __bf16* qb  = wob + (size_t)HID_ * H_ * D_;         // T*(H*D)
  __bf16* kb  = qb  + (size_t)T_ * H_ * D_;           // T*D
  __bf16* vb  = kb  + (size_t)T_ * D_;
  __bf16* ab  = vb  + (size_t)T_ * D_;                // T*(H*D)

  auto cvt = [&](const float* src, __bf16* dst, size_t n) {
    cvt_f32_bf16<<<(int)((n / 2 + 255) / 256), 256, 0, stream>>>(src, dst,
                                                                 (int)n);
  };
  cvt(hidden, hb,  (size_t)T_ * HID_);
  cvt(wq,     wqb, (size_t)H_ * D_ * HID_);
  cvt(wk,     wkb, (size_t)KV_ * D_ * HID_);
  cvt(wv,     wvb, (size_t)KV_ * D_ * HID_);
  cvt(wo,     wob, (size_t)HID_ * H_ * D_);

  auto gemm_grid = [&](int T, int N) { return (T >> 7) * (N >> 4); };

  // Q/K/V projections (bf16 out)
  gemm_bf16_wmma<<<gemm_grid(T_, H_ * D_), 256, 0, stream>>>(
      hb, wqb, qb, T_, H_ * D_, HID_, 1);
  gemm_bf16_wmma<<<gemm_grid(T_, KV_ * D_), 256, 0, stream>>>(
      hb, wkb, kb, T_, KV_ * D_, HID_, 1);
  gemm_bf16_wmma<<<gemm_grid(T_, KV_ * D_), 256, 0, stream>>>(
      hb, wvb, vb, T_, KV_ * D_, HID_, 1);

  // RoPE in place on bf16 q/k
  {
    int total = T_ * (H_ + KV_) * (D_ / 2);
    rope_kernel<<<(total + 255) / 256, 256, 0, stream>>>(qb, kb, pos);
  }

  // Flash attention -> bf16 context
  attn_kernel<<<B_ * H_ * (S_ / 16), 128, 0, stream>>>(qb, kb, vb, mask, ab);

  // Output projection (fp32 out)
  gemm_bf16_wmma<<<gemm_grid(T_, HID_), 256, 0, stream>>>(
      ab, wob, out, T_, HID_, H_ * D_, 0);
}
